// SimPGCN_37495064494301
// MI455X (gfx1250) — matrix operand order repaired and checked by reading, not compiled, not run on hardware
//
#include <hip/hip_runtime.h>
#include <hip/hip_bf16.h>
#include <math.h>

// ---------------------------------------------------------------------------
// SimPGCN forward, CDNA5 (gfx1250, wave32).
// GEMMs: v_wmma_f32_16x16x32_f16; B pre-packed into fragment layout (f16),
// async-staged to LDS in double-buffered 4-K-block groups
// (global_load_async_to_lds_b128 / s_wait_asynccnt, exact per-wave counts).
// SPMM: wave-per-edge L2-resident gather + f32 atomic scatter.
// ---------------------------------------------------------------------------

typedef __attribute__((ext_vector_type(16))) _Float16 v16h;
typedef __attribute__((ext_vector_type(8)))  float    v8f;

#define GAMMA 0.1f

// ---- CDNA5 async copy: 16 bytes global -> LDS, tracked on ASYNCcnt --------
__device__ __forceinline__ void async_copy_b128(unsigned lds_byte_addr,
                                                const void* gptr) {
    asm volatile("global_load_async_to_lds_b128 %0, %1, off"
                 :: "v"(lds_byte_addr),
                    "v"((unsigned long long)(uintptr_t)gptr)
                 : "memory");
}

template <int Ncnt>
__device__ __forceinline__ void wait_async() {
#if __has_builtin(__builtin_amdgcn_s_wait_asynccnt)
    __builtin_amdgcn_s_wait_asynccnt(Ncnt);
#else
    if constexpr (Ncnt == 0)       asm volatile("s_wait_asynccnt 0x0" ::: "memory");
    else if constexpr (Ncnt == 5)  asm volatile("s_wait_asynccnt 0x5" ::: "memory");
    else if constexpr (Ncnt == 13) asm volatile("s_wait_asynccnt 0xd" ::: "memory");
    else                           asm volatile("s_wait_asynccnt 0x0" ::: "memory");
#endif
}

// ---- pack B[K,Ncols] (f32, row-major) into WMMA B-fragment layout (f16) ---
// Fragment (kb32, j) = 512 halves; half index = ((kb32*NT + j)*32 + lane)*16 + e
// maps to B[kb32*32 + (lane>>4)*16 + e][j*16 + (lane&15)] (zero-padded cols).
__global__ void pack_B_frag_f16(const float* __restrict__ B,
                                _Float16* __restrict__ Bp,
                                int K, int Ncols, int NT) {
    const int idx = blockIdx.x * blockDim.x + threadIdx.x;
    const int total = (K >> 5) * NT * 512;
    if (idx >= total) return;
    const int e    = idx & 15;
    const int lane = (idx >> 4) & 31;
    const int frag = idx >> 9;
    const int j    = frag % NT;
    const int kb32 = frag / NT;
    const int k = (kb32 << 5) + ((lane >> 4) << 4) + e;
    const int n = (j << 4) + (lane & 15);
    const float v = (n < Ncols) ? B[(size_t)k * Ncols + n] : 0.0f;
    Bp[idx] = (_Float16)v;
}

// ---------------- WMMA GEMM: C[M,Ncols] = A[M,K] @ B[K,Ncols] --------------
// 5 waves / block (160 thr), each wave owns a 16-row strip of NT 16-col tiles.
// B fragments staged in double-buffered groups of CH=4 K-blocks; every thread
// issues exactly SI async b128 copies per group (chunk count padded to 160*SI),
// so "s_wait_asynccnt SI" precisely retires the previous group.
// Requires K % 128 == 0.
template <int NT, int SI>
__global__ __launch_bounds__(160) void
gemm_wmma_f16_lds(const float* __restrict__ A,
                  const _Float16* __restrict__ Bp,
                  float* __restrict__ Cout,
                  int M, int K, int Ncols) {
    constexpr int CH = 4;                        // K-blocks per group
    constexpr int CHUNKS_PAD = SI * 160;         // padded b128 chunks per group
    __shared__ __attribute__((aligned(32))) _Float16 sB[2][CHUNKS_PAD * 8];

    const int lane = threadIdx.x & 31;
    const int wv   = threadIdx.x >> 5;                 // 0..4
    const int nRowTiles = M >> 4;
    const int rowTile = blockIdx.x * 5 + wv;
    const bool active = rowTile < nRowTiles;
    const int rt = active ? rowTile : (nRowTiles - 1); // clamp for addr safety

    const int half = lane >> 4;
    const int l16  = lane & 15;
    const float* __restrict__ arow = A + (size_t)((rt << 4) + l16) * K;

    v8f zero = {};
    v8f acc[NT];
#pragma unroll
    for (int j = 0; j < NT; ++j) acc[j] = zero;

    const int nkb     = K >> 5;
    const int ngroups = nkb / CH;                // K % 128 == 0

    // issue group 0
    {
        const _Float16* src = Bp;
#pragma unroll
        for (int i = 0; i < SI; ++i) {
            const int c = threadIdx.x + i * 160;
            async_copy_b128((unsigned)(uintptr_t)&sB[0][c * 8], src + c * 8);
        }
    }

    for (int g = 0; g < ngroups; ++g) {
        const bool hasNext = (g + 1) < ngroups;
        if (hasNext) {  // stream next group behind this one (other buffer)
            const _Float16* src = Bp + (size_t)(g + 1) * (CH * NT * 512);
            _Float16* dst = sB[(g + 1) & 1];
#pragma unroll
            for (int i = 0; i < SI; ++i) {
                const int c = threadIdx.x + i * 160;
                async_copy_b128((unsigned)(uintptr_t)&dst[c * 8], src + c * 8);
            }
            wait_async<SI>();    // in-order: group g fully landed
        } else {
            wait_async<0>();
        }
        __syncthreads();         // group g visible to all waves

        const _Float16* buf = sB[g & 1];
#pragma unroll
        for (int kk = 0; kk < CH; ++kk) {
            const int kb32 = g * CH + kk;
            // ---- A fragment (16x32 f16, contiguous k = 16*half + e layout) ----
            const int kh = (kb32 << 5) + (half << 4);
            v16h a;
#pragma unroll
            for (int e = 0; e < 8; ++e) {
                a[e]     = (_Float16)arow[kh + e];
                a[8 + e] = (_Float16)arow[kh + 8 + e];
            }
#pragma unroll
            for (int j = 0; j < NT; ++j) {
                const v16h b = *reinterpret_cast<const v16h*>(
                    &buf[kk * NT * 512 + (j * 32 + lane) * 16]);
                acc[j] = __builtin_amdgcn_wmma_f32_16x16x32_f16(
                    false, a, false, b, (short)0, acc[j], false, false);
            }
        }
        if (hasNext)
            __builtin_prefetch(arow + ((g + 1) * CH) * 32, 0, 0); // global_prefetch_b8
        __syncthreads();         // all done reading sB[g&1] before g+2 overwrites
    }

    if (!active) return;
#pragma unroll
    for (int j = 0; j < NT; ++j) {
        const int n = (j << 4) + l16;
        if (n >= Ncols) continue;
#pragma unroll
        for (int r = 0; r < 8; ++r) {
            const int mm = (rowTile << 4) + r + (half << 3);
            Cout[(size_t)mm * Ncols + n] = acc[j][r];
        }
    }
}

// -------- SPMM scatter, D=128: wave per edge, float4 per lane --------------
__global__ void spmm_scatter_d128(const int* __restrict__ row,
                                  const int* __restrict__ col,
                                  const float* __restrict__ val,
                                  const float* __restrict__ X,
                                  float* __restrict__ out, int E) {
    const int wave = (blockIdx.x * (blockDim.x >> 5)) + (threadIdx.x >> 5);
    const int lane = threadIdx.x & 31;
    if (wave >= E) return;
    const int r = row[wave];
    const int c = col[wave];
    const float v = val[wave];
    const float4 x4 = ((const float4*)(X + (size_t)c * 128))[lane];
    float* o = out + (size_t)r * 128 + lane * 4;
    atomicAdd(o + 0, v * x4.x);
    atomicAdd(o + 1, v * x4.y);
    atomicAdd(o + 2, v * x4.z);
    atomicAdd(o + 3, v * x4.w);
}

// -------- SPMM scatter, D=40: wave per edge, lanes cover f and f+32 --------
__global__ void spmm_scatter_d40(const int* __restrict__ row,
                                 const int* __restrict__ col,
                                 const float* __restrict__ val,
                                 const float* __restrict__ X,
                                 float* __restrict__ out, int E) {
    const int wave = (blockIdx.x * (blockDim.x >> 5)) + (threadIdx.x >> 5);
    const int lane = threadIdx.x & 31;
    if (wave >= E) return;
    const int r = row[wave];
    const int c = col[wave];
    const float v = val[wave];
    const float* x = X + (size_t)c * 40;
    float* o = out + (size_t)r * 40;
    atomicAdd(o + lane, v * x[lane]);
    if (lane < 8) atomicAdd(o + lane + 32, v * x[lane + 32]);
}

// -------- per-row gates: s = sigmoid(x . sw + sb), Dk = x . dw + db --------
__global__ void row_gates(const float* __restrict__ X, int K,
                          const float* __restrict__ sw, const float* __restrict__ sb,
                          const float* __restrict__ dw, const float* __restrict__ db,
                          float* __restrict__ s_out, float* __restrict__ dk_out, int N) {
    const int wave = (blockIdx.x * (blockDim.x >> 5)) + (threadIdx.x >> 5);
    const int lane = threadIdx.x & 31;
    if (wave >= N) return;
    const float* xr = X + (size_t)wave * K;
    float as = 0.f, ad = 0.f;
    for (int f = lane; f < K; f += 32) {
        const float x = xr[f];
        as += x * sw[f];
        ad += x * dw[f];
    }
#pragma unroll
    for (int off = 16; off > 0; off >>= 1) {
        as += __shfl_xor(as, off, 32);
        ad += __shfl_xor(ad, off, 32);
    }
    if (lane == 0) {
        s_out[wave]  = 1.0f / (1.0f + __expf(-(as + sb[0])));
        dk_out[wave] = ad + db[0];
    }
}

// -------- layer-1 combine, D=128: one block of 128 threads per row ---------
__global__ void combine_d128(const float* __restrict__ hadj,
                             const float* __restrict__ hknn,
                             const float* __restrict__ sup,
                             const float* __restrict__ s,
                             const float* __restrict__ dk,
                             const float* __restrict__ bias,
                             float* __restrict__ out, int N) {
    const int row = blockIdx.x;
    if (row >= N) return;
    const int f = threadIdx.x;
    const size_t i = (size_t)row * 128 + f;
    const float b = bias[f];
    const float sv = s[row];
    const float dv = dk[row];
    out[i] = sv * (hadj[i] + b) + (1.0f - sv) * (hknn[i] + b)
           + GAMMA * dv * (sup[i] + b);
}

// -------- layer-2 combine + log_softmax, D=40: wave per row ----------------
__global__ void combine_logsoftmax_d40(const float* __restrict__ hadj,
                                       const float* __restrict__ hknn,
                                       const float* __restrict__ sup,
                                       const float* __restrict__ s,
                                       const float* __restrict__ dk,
                                       const float* __restrict__ bias,
                                       float* __restrict__ out, int N) {
    const int wave = (blockIdx.x * (blockDim.x >> 5)) + (threadIdx.x >> 5);
    const int lane = threadIdx.x & 31;
    if (wave >= N) return;
    const size_t base = (size_t)wave * 40;
    const float sv = s[wave];
    const float dv = dk[wave];

    const int f0 = lane;
    const int f1 = lane + 32;
    const float b0 = bias[f0];
    float z0 = sv * (hadj[base + f0] + b0) + (1.0f - sv) * (hknn[base + f0] + b0)
             + GAMMA * dv * (sup[base + f0] + b0);
    float z1 = -__builtin_inff();
    if (f1 < 40) {
        const float b1 = bias[f1];
        z1 = sv * (hadj[base + f1] + b1) + (1.0f - sv) * (hknn[base + f1] + b1)
           + GAMMA * dv * (sup[base + f1] + b1);
    }
    float zmax = fmaxf(z0, z1);
#pragma unroll
    for (int off = 16; off > 0; off >>= 1)
        zmax = fmaxf(zmax, __shfl_xor(zmax, off, 32));
    float acc = __expf(z0 - zmax) + ((f1 < 40) ? __expf(z1 - zmax) : 0.0f);
#pragma unroll
    for (int off = 16; off > 0; off >>= 1)
        acc += __shfl_xor(acc, off, 32);
    const float lse = zmax + __logf(acc);

    out[base + f0] = z0 - lse;
    if (f1 < 40) out[base + f1] = z1 - lse;
}

// ---------------------------------------------------------------------------
extern "C" void kernel_launch(void* const* d_in, const int* in_sizes, int n_in,
                              void* d_out, int out_size, void* d_ws, size_t ws_size,
                              hipStream_t stream) {
    const float* fea     = (const float*)d_in[0];
    const int*   adj_row = (const int*)d_in[1];
    const int*   adj_col = (const int*)d_in[2];
    const float* adj_val = (const float*)d_in[3];
    const int*   knn_row = (const int*)d_in[4];
    const int*   knn_col = (const int*)d_in[5];
    const float* knn_val = (const float*)d_in[6];
    const float* W1      = (const float*)d_in[7];
    const float* b1      = (const float*)d_in[8];
    const float* W2      = (const float*)d_in[9];
    const float* b2      = (const float*)d_in[10];
    const float* score1  = (const float*)d_in[11];
    const float* score2  = (const float*)d_in[12];
    const float* sbias1  = (const float*)d_in[13];
    const float* sbias2  = (const float*)d_in[14];
    const float* Dk1     = (const float*)d_in[15];
    const float* Dk2     = (const float*)d_in[16];
    const float* Dbias1  = (const float*)d_in[17];
    const float* Dbias2  = (const float*)d_in[18];

    const int F = 512, H = 128, C = 40;
    const int N  = in_sizes[0] / F;   // 50000
    const int E  = in_sizes[1];       // 800000
    const int EK = in_sizes[4];       // 1000000

    // workspace layout (floats, then f16 packed-B regions with pad slack)
    float* ws = (float*)d_ws;
    size_t o = 0;
    float* sup1  = ws + o; o += (size_t)N * H;
    float* hadj1 = ws + o; o += (size_t)N * H;
    float* hknn1 = ws + o; o += (size_t)N * H;
    float* sup2  = ws + o; o += (size_t)N * C;
    float* hadj2 = ws + o; o += (size_t)N * C;
    float* hknn2 = ws + o; o += (size_t)N * C;
    float* s1  = ws + o; o += N;
    float* dk1 = ws + o; o += N;
    float* s2  = ws + o; o += N;
    float* dk2 = ws + o; o += N;
    const int bp1_halves = (F >> 5) * 8 * 512;   // 65536
    const int bp2_halves = (H >> 5) * 3 * 512;   // 6144
    _Float16* bp1 = (_Float16*)(ws + o); o += (bp1_halves + 512) / 2; // +pad slack
    _Float16* bp2 = (_Float16*)(ws + o); o += (bp2_halves + 512) / 2;

    float* logits = (float*)d_out;              // [N,40]
    float* emb    = logits + (size_t)N * C;     // [N,128] layer-1 output

    hipMemsetAsync(hadj1, 0, (size_t)N * H * 2 * sizeof(float), stream);
    hipMemsetAsync(hadj2, 0, (size_t)N * C * 2 * sizeof(float), stream);

    const dim3 blk(256);
    const int wavesPerBlk = 256 / 32;
    const int rowTiles   = N >> 4;                         // 3125
    const int gemmBlocks = (rowTiles + 4) / 5;             // 625 (5 waves/block)
    const int rowBlocks  = (N + wavesPerBlk - 1) / wavesPerBlk;

    // pre-pack both weight matrices into WMMA fragment layout (f16)
    pack_B_frag_f16<<<(bp1_halves + 255) / 256, blk, 0, stream>>>(W1, bp1, F, H, 8);
    pack_B_frag_f16<<<(bp2_halves + 255) / 256, blk, 0, stream>>>(W2, bp2, H, C, 3);

    // ---- layer 1 ----  (NT=8; group = 4 K-blocks = 2048 chunks -> SI=13)
    gemm_wmma_f16_lds<8, 13><<<gemmBlocks, dim3(160), 0, stream>>>(fea, bp1, sup1, N, F, H);
    row_gates<<<rowBlocks, blk, 0, stream>>>(fea, F, score1, sbias1, Dk1, Dbias1, s1, dk1, N);
    spmm_scatter_d128<<<(E  + wavesPerBlk - 1) / wavesPerBlk, blk, 0, stream>>>(
        adj_row, adj_col, adj_val, sup1, hadj1, E);
    spmm_scatter_d128<<<(EK + wavesPerBlk - 1) / wavesPerBlk, blk, 0, stream>>>(
        knn_row, knn_col, knn_val, sup1, hknn1, EK);
    combine_d128<<<N, dim3(128), 0, stream>>>(hadj1, hknn1, sup1, s1, dk1, b1, emb, N);

    // ---- layer 2 ----  (NT=3; group = 4 K-blocks = 768 chunks -> SI=5)
    gemm_wmma_f16_lds<3, 5><<<gemmBlocks, dim3(160), 0, stream>>>(emb, bp2, sup2, N, H, C);
    row_gates<<<rowBlocks, blk, 0, stream>>>(emb, H, score2, sbias2, Dk2, Dbias2, s2, dk2, N);
    spmm_scatter_d40<<<(E  + wavesPerBlk - 1) / wavesPerBlk, blk, 0, stream>>>(
        adj_row, adj_col, adj_val, sup2, hadj2, E);
    spmm_scatter_d40<<<(EK + wavesPerBlk - 1) / wavesPerBlk, blk, 0, stream>>>(
        knn_row, knn_col, knn_val, sup2, hknn2, EK);
    combine_logsoftmax_d40<<<rowBlocks, blk, 0, stream>>>(
        hadj2, hknn2, sup2, s2, dk2, b2, logits, N);
}
